// CSWinBlock3D_30588757082813
// MI455X (gfx1250) — matrix operand order repaired and checked by reading, not compiled, not run on hardware
//
#include <hip/hip_runtime.h>
#include <math.h>

// ---------------- problem constants ----------------
#define L_TOK 32768   // 32*32*32 tokens
#define C_DIM 512
#define QKV_N 1536
#define HID_N 2048

typedef __attribute__((ext_vector_type(16))) __bf16 v16bf;
typedef __attribute__((ext_vector_type(8)))  __bf16 bf16x8;
typedef __attribute__((ext_vector_type(8)))  float  v8f;
typedef __attribute__((ext_vector_type(4)))  unsigned int u32x4;
typedef __attribute__((ext_vector_type(8)))  unsigned int u32x8;

__device__ __forceinline__ __bf16 f2bf(float f) {
  unsigned u = __builtin_bit_cast(unsigned, f);
  unsigned r = (u + 0x7FFFu + ((u >> 16) & 1u)) >> 16;   // round-to-nearest-even
  return __builtin_bit_cast(__bf16, (unsigned short)r);
}
__device__ __forceinline__ float bf2f(__bf16 h) {
  unsigned u = ((unsigned)__builtin_bit_cast(unsigned short, h)) << 16;
  return __builtin_bit_cast(float, u);
}
__device__ __forceinline__ v16bf cat8(bf16x8 lo, bf16x8 hi) {
  return __builtin_shufflevector(lo, hi, 0,1,2,3,4,5,6,7,8,9,10,11,12,13,14,15);
}
__device__ __forceinline__ v8f wmma_bf16(v16bf a, v16bf b, v8f c) {
  return __builtin_amdgcn_wmma_f32_16x16x32_bf16(false, a, false, b, (short)0, c, false, false);
}
__device__ __forceinline__ float gelu_exact(float x) {
  return 0.5f * x * (1.0f + erff(x * 0.70710678118654752f));
}

// LDS offset of a __shared__ object = low 32 bits of its generic address
// (per the ISA's aperture rule).
__device__ __forceinline__ unsigned lds_off(const void* p) {
  return (unsigned)(size_t)p;
}

// ---- Tensor Data Mover: 2D tile load Global->LDS (2-byte elems, LDS padding)
// pad_interval=16 DWORDs (one 32-elem bf16 row), pad_amount=4 DWORDs (8 bf16)
// => effective LDS row stride = 40 bf16 (== LDA_P below).
__device__ __forceinline__ void tdm_load_2d_bf16(unsigned ldsaddr, const void* gp,
                                                 unsigned tile0, unsigned tile1,
                                                 unsigned stride0) {
  unsigned long long ga = (unsigned long long)(size_t)gp;
  u32x4 g0;
  g0[0] = 1u;                                   // count=1, user mode
  g0[1] = ldsaddr;                              // lds_addr
  g0[2] = (unsigned)ga;                         // global_addr[31:0]
  g0[3] = ((unsigned)(ga >> 32) & 0x01FFFFFFu)  // global_addr[56:32]
          | 0x80000000u;                        // type=2 ("image")
  u32x8 g1;
  g1[0] = (1u << 16)        // data_size = 2 bytes
        | (1u << 20)        // pad_enable
        | (3u << 22)        // pad_interval: 16 DWORDs
        | (3u << 25);       // pad_amount:   4 DWORDs
  g1[1] = (tile0 & 0xFFFFu) << 16;              // tensor_dim0[15:0] @bit48
  g1[2] = (tile0 >> 16) | ((tile1 & 0xFFFFu) << 16);   // tdim0 hi | tdim1 lo
  g1[3] = (tile1 >> 16) | (tile0 << 16);        // tdim1 hi | tile_dim0
  g1[4] = tile1;                                // tile_dim1 | tile_dim2=0
  g1[5] = stride0;                              // tensor_dim0_stride[31:0]
  g1[6] = 0;                                    // stride hi | dim1_stride lo
  g1[7] = 0;
  asm volatile("tensor_load_to_lds %0, %1" :: "s"(g0), "s"(g1) : "memory");
}

// ---- async Global->LDS 16B copy (per-lane), tracked by ASYNCcnt ------------
__device__ __forceinline__ void async_ld_b128(unsigned ldsaddr, const void* gp) {
  asm volatile("global_load_async_to_lds_b128 %0, %1, off"
               :: "v"(ldsaddr), "v"((unsigned long long)(size_t)gp) : "memory");
}
__device__ __forceinline__ void wait_asynccnt0() {
  asm volatile("s_wait_asynccnt 0x0" ::: "memory");
}

// ---------------- fp32 -> bf16 convert (weights) ----------------
__global__ __launch_bounds__(256) void cvt_kernel(const float* __restrict__ in,
                                                  __bf16* __restrict__ out, int n) {
  int i = blockIdx.x * 256 + threadIdx.x;
  if (i < n) out[i] = f2bf(in[i]);
}

// ---------------- LayerNorm over C=512, output bf16 (L,C) ----------------
__global__ __launch_bounds__(256)
void layernorm_bf16_kernel(const float* __restrict__ in, const float* __restrict__ g,
                           const float* __restrict__ b, __bf16* __restrict__ out,
                           int cmajor) {
  const int C = C_DIM;
  int t = blockIdx.x;
  int c0 = threadIdx.x, c1 = threadIdx.x + 256;
  float v0, v1;
  if (cmajor) { v0 = in[(size_t)c0 * L_TOK + t]; v1 = in[(size_t)c1 * L_TOK + t]; }
  else        { v0 = in[(size_t)t * C + c0];     v1 = in[(size_t)t * C + c1]; }
  __shared__ float red[256];
  red[threadIdx.x] = v0 + v1;
  __syncthreads();
  for (int o = 128; o > 0; o >>= 1) {
    if (threadIdx.x < o) red[threadIdx.x] += red[threadIdx.x + o];
    __syncthreads();
  }
  float mean = red[0] * (1.0f / 512.0f);
  __syncthreads();
  float d0 = v0 - mean, d1 = v1 - mean;
  red[threadIdx.x] = d0 * d0 + d1 * d1;
  __syncthreads();
  for (int o = 128; o > 0; o >>= 1) {
    if (threadIdx.x < o) red[threadIdx.x] += red[threadIdx.x + o];
    __syncthreads();
  }
  float inv = rsqrtf(red[0] * (1.0f / 512.0f) + 1e-5f);
  out[(size_t)t * C + c0] = f2bf(d0 * inv * g[c0] + b[c0]);
  out[(size_t)t * C + c1] = f2bf(d1 * inv * g[c1] + b[c1]);
}

// ---------------- bf16 WMMA GEMM: C(MxN) = A(MxK) * B(KxN) [+epilogue] ----
// Block tile 128x64, 8 waves; A tile staged by the Tensor Data Mover.
// EPI: 0 plain, 1 +bias, 2 +bias+GELU, 3 +bias+resid(LxN), 4 +bias+resid(NxM)
#define LDA_P 40   // padded LDS row stride (bf16) — matches TDM pad settings
template <typename OutT, int EPI>
__global__ __launch_bounds__(256)
void gemm_bf16_kernel(const __bf16* __restrict__ A, const __bf16* __restrict__ Bm,
                      const float* __restrict__ bias, const float* __restrict__ resid,
                      OutT* __restrict__ Cout, int M, int N, int K) {
  __shared__ __bf16 sA[128][LDA_P];
  __shared__ __bf16 sB[64][LDA_P];
  const int m0 = blockIdx.y * 128;
  const int n0 = blockIdx.x * 64;
  const int tid = threadIdx.x;
  const int lane = tid & 31, wave = tid >> 5;
  const int wm = wave & 3, wn = wave >> 2;          // 4x2 wave grid
  const int laneR = lane & 15, laneH = lane >> 4;
  const unsigned sA_base = lds_off(&sA[0][0]);

  v8f acc[2][2] = {};

  for (int k0 = 0; k0 < K; k0 += 32) {
    __syncthreads();
    // A tile 128x32 via TDM (one descriptor, wave 0 issues; EXEC-independent)
    if (wave == 0)
      tdm_load_2d_bf16(sA_base, A + (size_t)m0 * K + k0, 32u, 128u, (unsigned)K);
    {   // stage B tile 32x64, transposed to (n,k) in LDS (overlaps TDM)
      int k = tid >> 3, nc = tid & 7;
      const __bf16* src = Bm + (size_t)(k0 + k) * N + n0 + nc * 8;
      bf16x8 v = *(const bf16x8*)src;
      if (k0 + 32 < K) __builtin_prefetch(src + (size_t)32 * N, 0, 1);
#pragma unroll
      for (int i = 0; i < 8; ++i) sB[nc * 8 + i][k] = v[i];
    }
    if (wave == 0) __builtin_amdgcn_s_wait_tensorcnt(0);
    __syncthreads();

    v16bf af[2], bf[2];
#pragma unroll
    for (int r = 0; r < 2; ++r) {   // A frag: 16-bit 16x32 ISA layout
      int row = wm * 32 + r * 16 + laneR;
      bf16x8 lo = *(const bf16x8*)&sA[row][laneH * 8];
      bf16x8 hi = *(const bf16x8*)&sA[row][16 + laneH * 8];
      af[r] = cat8(lo, hi);
    }
#pragma unroll
    for (int c = 0; c < 2; ++c) {   // B frag: 16-bit 32x16 ISA layout
      int col = wn * 32 + c * 16 + laneR;
      bf16x8 lo = *(const bf16x8*)&sB[col][laneH * 8];
      bf16x8 hi = *(const bf16x8*)&sB[col][16 + laneH * 8];
      bf[c] = cat8(lo, hi);
    }
#pragma unroll
    for (int r = 0; r < 2; ++r)
#pragma unroll
      for (int c = 0; c < 2; ++c)
        acc[r][c] = wmma_bf16(af[r], bf[c], acc[r][c]);
  }

#pragma unroll
  for (int r = 0; r < 2; ++r)
#pragma unroll
    for (int c = 0; c < 2; ++c) {
      int row0 = m0 + wm * 32 + r * 16 + laneH * 8;
      int col  = n0 + wn * 32 + c * 16 + laneR;
      float bs = (EPI >= 1) ? bias[col] : 0.0f;
#pragma unroll
      for (int j = 0; j < 8; ++j) {
        int row = row0 + j;
        float v = acc[r][c][j] + bs;
        if constexpr (EPI == 2) v = gelu_exact(v);
        if constexpr (EPI == 3) v += resid[(size_t)row * N + col];
        if constexpr (EPI == 4) v += resid[(size_t)col * M + row];
        OutT* p = Cout + (size_t)row * N + col;
        if constexpr (__is_same(OutT, float)) *p = v; else *p = f2bf(v);
      }
    }
}

// ---------------- CSWin window attention: one block per (branch,window,head) -
__global__ __launch_bounds__(256)
void attn_kernel(const __bf16* __restrict__ qkv, __bf16* __restrict__ att) {
  int bw = blockIdx.x;
  int branch = bw >> 11;
  int id = bw & 2047;
  int widx = id >> 3;
  int head = id & 7;
  int d = widx >> 3, sub = widx & 7;           // sub = wb (b0) or hb (b1)
  int cb = branch * 256 + head * 32;

  __shared__ union {
    struct { __bf16 q[128][40]; __bf16 k[128][40]; } qk;
    __bf16 p[128][136];
  } u;
  __shared__ __bf16 vt[32][136];               // V transposed: [channel][token]

  int tid = threadIdx.x;
  auto tok = [&](int p) -> int {
    return branch ? d * 1024 + (sub * 4 + (p >> 5)) * 32 + (p & 31)
                  : d * 1024 + (p >> 2) * 32 + sub * 4 + (p & 3);
  };

  {   // K via async LDS loads; Q (scaled) and V^T manually
    int p = tid >> 1, half = tid & 1;
    const __bf16* base = qkv + (size_t)tok(p) * QKV_N + cb + half * 16;
    unsigned koff = lds_off(&u.qk.k[p][half * 16]);
    async_ld_b128(koff,      base + 512);
    async_ld_b128(koff + 16, base + 520);
#pragma unroll
    for (int i = 0; i < 16; ++i)
      u.qk.q[p][half * 16 + i] = f2bf(bf2f(base[i]) * 0.17677669529663687f);
#pragma unroll
    for (int i = 0; i < 16; ++i)
      vt[half * 16 + i][p] = base[1024 + i];
  }
  wait_asynccnt0();
  __syncthreads();

  int lane = tid & 31, wave = tid >> 5;
  int laneR = lane & 15, laneH = lane >> 4;

  // S = Q*K^T : wave owns 16 rows x 128 cols (8 col tiles), single K-step (32)
  v16bf aq;
  {
    int row = wave * 16 + laneR;
    aq = cat8(*(const bf16x8*)&u.qk.q[row][laneH * 8],
              *(const bf16x8*)&u.qk.q[row][16 + laneH * 8]);
  }
  v8f s[8];
#pragma unroll
  for (int c = 0; c < 8; ++c) {
    int col = c * 16 + laneR;
    v16bf bk = cat8(*(const bf16x8*)&u.qk.k[col][laneH * 8],
                    *(const bf16x8*)&u.qk.k[col][16 + laneH * 8]);
    v8f z = {};
    s[c] = wmma_bf16(aq, bk, z);
  }

  // softmax along each row (row = wave*16 + j + 8*laneH, spread across 16 lanes)
  float rsum[8];
#pragma unroll
  for (int j = 0; j < 8; ++j) {
    float m = s[0][j];
#pragma unroll
    for (int c = 1; c < 8; ++c) m = fmaxf(m, s[c][j]);
    m = fmaxf(m, __shfl_xor(m, 1, 32));
    m = fmaxf(m, __shfl_xor(m, 2, 32));
    m = fmaxf(m, __shfl_xor(m, 4, 32));
    m = fmaxf(m, __shfl_xor(m, 8, 32));
    float sum = 0.0f;
#pragma unroll
    for (int c = 0; c < 8; ++c) { float e = __expf(s[c][j] - m); s[c][j] = e; sum += e; }
    sum += __shfl_xor(sum, 1, 32);
    sum += __shfl_xor(sum, 2, 32);
    sum += __shfl_xor(sum, 4, 32);
    sum += __shfl_xor(sum, 8, 32);
    rsum[j] = 1.0f / sum;
  }

  __syncthreads();   // all Q/K LDS reads complete before P overwrites the union
#pragma unroll
  for (int c = 0; c < 8; ++c)
#pragma unroll
    for (int j = 0; j < 8; ++j)
      u.p[wave * 16 + j + 8 * laneH][c * 16 + laneR] = f2bf(s[c][j] * rsum[j]);
  __syncthreads();

  // O = P*V : rows per wave 16, N=32, K=128 -> 4 K-steps x 2 col tiles
  v8f o[2] = {};
#pragma unroll
  for (int kt = 0; kt < 4; ++kt) {
    int row = wave * 16 + laneR;
    v16bf ap = cat8(*(const bf16x8*)&u.p[row][kt * 32 + laneH * 8],
                    *(const bf16x8*)&u.p[row][kt * 32 + 16 + laneH * 8]);
#pragma unroll
    for (int c = 0; c < 2; ++c) {
      int ch = c * 16 + laneR;
      v16bf bv = cat8(*(const bf16x8*)&vt[ch][kt * 32 + laneH * 8],
                      *(const bf16x8*)&vt[ch][kt * 32 + 16 + laneH * 8]);
      o[c] = wmma_bf16(ap, bv, o[c]);
    }
  }

#pragma unroll
  for (int c = 0; c < 2; ++c)
#pragma unroll
    for (int j = 0; j < 8; ++j) {
      int p = wave * 16 + j + 8 * laneH;
      int ch = c * 16 + laneR;
      att[(size_t)tok(p) * C_DIM + cb + ch] = f2bf(o[c][j]);
    }
}

// ---------------- LEPE depthwise 3x3 (per-window SAME pad), accumulate -------
__global__ __launch_bounds__(256)
void lepe_kernel(const __bf16* __restrict__ qkv,
                 const float* __restrict__ w0, const float* __restrict__ b0,
                 const float* __restrict__ w1, const float* __restrict__ b1,
                 __bf16* __restrict__ att) {
  int i = blockIdx.x * 256 + threadIdx.x;     // 2*256*256*128 total
  int p = i & 127;
  int c = (i >> 7) & 255;
  int widx = (i >> 15) & 255;
  int branch = i >> 23;
  int d = widx >> 3, sub = widx & 7;
  const float* W = (branch ? w1 : w0) + c * 9;
  float acc = branch ? b1[c] : b0[c];
  int Hs = branch ? 4 : 32, Ws = branch ? 32 : 4;
  int hs = branch ? (p >> 5) : (p >> 2);
  int ws = branch ? (p & 31) : (p & 3);
#pragma unroll
  for (int di = -1; di <= 1; ++di)
#pragma unroll
    for (int dj = -1; dj <= 1; ++dj) {
      int hh = hs + di, ww = ws + dj;
      if (hh < 0 || hh >= Hs || ww < 0 || ww >= Ws) continue;
      int t = branch ? d * 1024 + (sub * 4 + hh) * 32 + ww
                     : d * 1024 + hh * 32 + sub * 4 + ww;
      acc += W[(di + 1) * 3 + (dj + 1)] *
             bf2f(qkv[(size_t)t * QKV_N + 1024 + branch * 256 + c]);
    }
  int t0 = branch ? d * 1024 + (sub * 4 + hs) * 32 + ws
                  : d * 1024 + hs * 32 + sub * 4 + ws;
  size_t oi = (size_t)t0 * C_DIM + branch * 256 + c;
  att[oi] = f2bf(bf2f(att[oi]) + acc);
}

// ---------------- (L,C) f32 -> (C,L) f32 output transpose --------------------
__global__ __launch_bounds__(256)
void transpose_kernel(const float* __restrict__ in, float* __restrict__ out) {
  int i = blockIdx.x * 256 + threadIdx.x;     // 16777216 total
  int t = i >> 9, c = i & 511;
  out[(size_t)c * L_TOK + t] = in[i];
}

// ---------------- host launcher ----------------------------------------------
extern "C" void kernel_launch(void* const* d_in, const int* in_sizes, int n_in,
                              void* d_out, int out_size, void* d_ws, size_t ws_size,
                              hipStream_t stream) {
  const float* x      = (const float*)d_in[0];
  const float* n1g    = (const float*)d_in[1];
  const float* n1b    = (const float*)d_in[2];
  const float* qkv_w  = (const float*)d_in[3];
  const float* l0w    = (const float*)d_in[4];
  const float* l0b    = (const float*)d_in[5];
  const float* l1w    = (const float*)d_in[6];
  const float* l1b    = (const float*)d_in[7];
  const float* proj_w = (const float*)d_in[8];
  const float* proj_b = (const float*)d_in[9];
  const float* n2g    = (const float*)d_in[10];
  const float* n2b    = (const float*)d_in[11];
  const float* fc1_w  = (const float*)d_in[12];
  const float* fc1_b  = (const float*)d_in[13];
  const float* fc2_w  = (const float*)d_in[14];
  const float* fc2_b  = (const float*)d_in[15];
  float* out = (float*)d_out;

  char* ws = (char*)d_ws;
  size_t off = 0;
  auto carve = [&](size_t bytes) -> char* {
    char* p = ws + off;
    off = (off + bytes + 255) & ~(size_t)255;
    return p;
  };
  __bf16* wqkv = (__bf16*)carve((size_t)C_DIM * QKV_N * 2);
  __bf16* wprj = (__bf16*)carve((size_t)C_DIM * C_DIM * 2);
  __bf16* wfc1 = (__bf16*)carve((size_t)C_DIM * HID_N * 2);
  __bf16* wfc2 = (__bf16*)carve((size_t)HID_N * C_DIM * 2);
  __bf16* xln  = (__bf16*)carve((size_t)L_TOK * C_DIM * 2);
  __bf16* qkvb = (__bf16*)carve((size_t)L_TOK * QKV_N * 2);
  __bf16* attb = (__bf16*)carve((size_t)L_TOK * C_DIM * 2);
  float*  xf   = (float*) carve((size_t)L_TOK * C_DIM * 4);
  __bf16* h2   = (__bf16*)carve((size_t)L_TOK * C_DIM * 2);
  __bf16* hidb = (__bf16*)carve((size_t)L_TOK * HID_N * 2);
  float*  tmp  = (float*) carve((size_t)L_TOK * C_DIM * 4);

  cvt_kernel<<<(C_DIM * QKV_N + 255) / 256, 256, 0, stream>>>(qkv_w, wqkv, C_DIM * QKV_N);
  cvt_kernel<<<(C_DIM * C_DIM + 255) / 256, 256, 0, stream>>>(proj_w, wprj, C_DIM * C_DIM);
  cvt_kernel<<<(C_DIM * HID_N + 255) / 256, 256, 0, stream>>>(fc1_w, wfc1, C_DIM * HID_N);
  cvt_kernel<<<(HID_N * C_DIM + 255) / 256, 256, 0, stream>>>(fc2_w, wfc2, HID_N * C_DIM);

  layernorm_bf16_kernel<<<L_TOK, 256, 0, stream>>>(x, n1g, n1b, xln, 1);

  gemm_bf16_kernel<__bf16, 0><<<dim3(QKV_N / 64, L_TOK / 128), 256, 0, stream>>>(
      xln, wqkv, nullptr, nullptr, qkvb, L_TOK, QKV_N, C_DIM);

  attn_kernel<<<4096, 256, 0, stream>>>(qkvb, attb);

  lepe_kernel<<<(2 * 256 * 256 * 128) / 256, 256, 0, stream>>>(qkvb, l0w, l0b, l1w, l1b, attb);

  gemm_bf16_kernel<float, 4><<<dim3(C_DIM / 64, L_TOK / 128), 256, 0, stream>>>(
      attb, wprj, proj_b, x, xf, L_TOK, C_DIM, C_DIM);

  layernorm_bf16_kernel<<<L_TOK, 256, 0, stream>>>(xf, n2g, n2b, h2, 0);

  gemm_bf16_kernel<__bf16, 2><<<dim3(HID_N / 64, L_TOK / 128), 256, 0, stream>>>(
      h2, wfc1, fc1_b, nullptr, hidb, L_TOK, HID_N, C_DIM);

  gemm_bf16_kernel<float, 3><<<dim3(C_DIM / 64, L_TOK / 128), 256, 0, stream>>>(
      hidb, wfc2, fc2_b, xf, tmp, L_TOK, C_DIM, HID_N);

  transpose_kernel<<<(L_TOK * C_DIM) / 256, 256, 0, stream>>>(tmp, out);
}
